// GNNHypernetwork_10677288698538
// MI455X (gfx1250) — matrix operand order repaired and checked by reading, not compiled, not run on hardware
//
#include <hip/hip_runtime.h>
#include <hip/hip_bf16.h>
#include <math.h>
#include <stdint.h>

// ---------------------------------------------------------------------------
// Problem constants (from the reference)
// ---------------------------------------------------------------------------
#define BB   64          // batch
#define PP   50          // nodes per graph
#define MM   512         // raw feature dim
#define HH   64          // hidden unit
#define EE   400         // edges per graph
#define NEG_SLOPE 0.01f

// encoder dims
#define K1 (PP*MM)       // 25600
#define N1 (PP*HH)       // 3200
#define N2 (2*PP*HH)     // 6400
#define N3 (4*PP*HH)     // 12800

// GCN dims
#define FIN  (4*HH)      // 256
#define FH   (2*HH)      // 128
#define NEDGE (EE+PP)    // 450 (with self loops)
#define ZLEN 2000        // 1000*2

#define KCHUNK 64        // K rows per TDM tile (2 WMMA k-steps)

typedef __attribute__((ext_vector_type(16))) __bf16 v16bf;
typedef __attribute__((ext_vector_type(8)))  float  v8f;
typedef __attribute__((ext_vector_type(4)))  unsigned int u32x4;
typedef __attribute__((ext_vector_type(8)))  int    i32x8;
typedef __attribute__((ext_vector_type(4)))  int    i32x4;

union Frag32B { v16bf bf; unsigned int u[8]; };

// fp32 -> bf16 round-to-nearest-even (used in cold paths)
__device__ __forceinline__ unsigned short f32_to_bf16(float f) {
    unsigned int u = __builtin_bit_cast(unsigned int, f);
    u = u + 0x7fffu + ((u >> 16) & 1u);
    return (unsigned short)(u >> 16);
}
__device__ __forceinline__ float leaky(float v) {
    return (v >= 0.f) ? v : NEG_SLOPE * v;
}

// ---------------------------------------------------------------------------
// Tensor Data Mover: load a KCHUNK x 32 fp32 tile of W (row stride N) to LDS.
// D# per CDNA5 ISA ch.8: group0 = {count/lds_addr/global_addr/type},
// group1 = {data_size, tensor_dim0/1, tile_dim0/1, tensor_dim0_stride}.
// ---------------------------------------------------------------------------
__device__ __forceinline__ void tdm_load_tile(const float* gsrc, unsigned lds_off,
                                              int N, int K) {
    unsigned long long ga = (unsigned long long)(uintptr_t)gsrc;
    u32x4 g0;
    g0.x = 1u;                                                  // count=1 (valid)
    g0.y = lds_off;                                             // lds_addr (bytes)
    g0.z = (unsigned)ga;                                        // global_addr[31:0]
    g0.w = (unsigned)((ga >> 32) & 0x01ffffffu) | 0x80000000u;  // [56:32] | type=2

    i32x8 g1;
    g1[0] = 2 << 16;                                            // data_size = 4B
    g1[1] = (int)(((unsigned)N & 0xffffu) << 16);               // tensor_dim0 lo @ [63:48]
    g1[2] = (int)((((unsigned)N >> 16) & 0xffffu) |
                  (((unsigned)K & 0xffffu) << 16));             // dim0 hi | tensor_dim1 lo
    g1[3] = (int)((((unsigned)K >> 16) & 0xffffu) |
                  (32u << 16));                                 // dim1 hi | tile_dim0 = 32
    g1[4] = KCHUNK;                                             // tile_dim1 = 64, tile_dim2 = 0
    g1[5] = N;                                                  // tensor_dim0_stride[31:0]
    g1[6] = 0;                                                  // stride hi | dim1_stride lo
    g1[7] = 0;

    i32x4 gz4 = {0, 0, 0, 0};
#if defined(__clang_major__) && __clang_major__ >= 23
    i32x8 gz8 = {0, 0, 0, 0, 0, 0, 0, 0};
    __builtin_amdgcn_tensor_load_to_lds(g0, g1, gz4, gz4, gz8, 0);
#else
    __builtin_amdgcn_tensor_load_to_lds(g0, g1, gz4, gz4, 0);
#endif
}

// ---------------------------------------------------------------------------
// Kernel 1: fp32 -> bf16 elementwise (activation pre-convert)
// ---------------------------------------------------------------------------
__global__ __launch_bounds__(256)
void cvt_f32_bf16(const float* __restrict__ src, unsigned short* __restrict__ dst, int n) {
    int i = blockIdx.x * 256 + threadIdx.x;
    if (i < n) dst[i] = f32_to_bf16(src[i]);
}

// ---------------------------------------------------------------------------
// Kernel 2: weight-streaming GEMM, M = 64 fixed.
//   C(64,N) = A(64,K)[bf16] * W(K,N)[fp32 -> bf16 in-reg] + bias
//   grid.x = N/32 ; block = 256 threads = 8 waves.
//   wave w: M-tile = w&3, N-sub = w>>2; one 16x16 C tile per wave.
//   W tiles (64K x 32N fp32) streamed into LDS by the Tensor Data Mover,
//   double-buffered, synchronized with s_wait_tensorcnt + barriers.
// ---------------------------------------------------------------------------
template<bool WRITE_F32, bool WRITE_BF16, bool LEAKY>
__global__ __launch_bounds__(256)
void gemm_bf16(const unsigned short* __restrict__ A,   // (64,K) bf16 row-major
               const float*          __restrict__ W,   // (K,N)  f32 row-major
               const float*          __restrict__ bias,// (N)
               float*                __restrict__ outF, // (64,N) f32
               unsigned short*       __restrict__ outB, // (64,N) bf16
               int K, int N)
{
    __shared__ __align__(16) float sW[2][KCHUNK * 32];  // 2 x 8KB double buffer

    const int tid  = threadIdx.x;
    const int wave = tid >> 5;
    const int lane = tid & 31;
    const int h    = lane >> 4;           // lane half
    const int lr   = lane & 15;
    const int mt   = wave & 3;            // M tile (0..3)
    const int nsub = wave >> 2;           // 0/1: which 16-col half of the 32-col strip
    const int col0 = blockIdx.x * 32;     // block's first output column
    const int n    = col0 + nsub * 16 + lr;  // this lane's output column
    const int ncol = nsub * 16 + lr;         // column within LDS tile
    const int mrow = mt * 16 + lr;           // this lane's A row

    // A-fragment k offsets for 16-bit A 16x32 layout (ISA 7.12.2)
    int akoff[8];
#pragma unroll
    for (int v = 0; v < 8; ++v)
        akoff[v] = (v < 4) ? (h * 8 + 2 * v) : (16 + h * 8 + 2 * (v - 4));

    const unsigned short* Arow = A + (size_t)mrow * K;
    const unsigned lds0 = (unsigned)(uintptr_t)(&sW[0][0]);
    const unsigned lds1 = (unsigned)(uintptr_t)(&sW[1][0]);

    const int nchunks = K / KCHUNK;

    // prologue: DMA chunk 0 into buffer 0
    if (wave == 0)
        tdm_load_tile(W + col0, lds0, N, K);

    v8f acc = {};

    for (int c = 0; c < nchunks; ++c) {
        const int cur = c & 1;
        if (wave == 0) {
            if (c + 1 < nchunks) {
                tdm_load_tile(W + (size_t)(c + 1) * KCHUNK * N + col0,
                              cur ? lds0 : lds1, N, K);
                __builtin_amdgcn_s_wait_tensorcnt(1);  // chunk c resident
            } else {
                __builtin_amdgcn_s_wait_tensorcnt(0);
            }
        }
        __syncthreads();                               // publish buffer `cur`

        // prefetch A two chunks ahead (global_prefetch_b8)
        if ((c + 2) * KCHUNK < K)
            __builtin_prefetch(Arow + (size_t)(c + 2) * KCHUNK, 0, 3);

        volatile const float* sWc = &sW[cur][0];       // TDM-written; don't elide
#pragma unroll
        for (int ks = 0; ks < 2; ++ks) {
            Frag32B a, b;
            const unsigned short* Ak = Arow + (size_t)c * KCHUNK + ks * 32;
#pragma unroll
            for (int v = 0; v < 8; ++v)
                a.u[v] = *(const unsigned int*)(Ak + akoff[v]);

            // B 32x16 bf16 layout: same k order as A (v<4: K=8h+2v; v>=4: K=16+8h+2(v-4))
#pragma unroll
            for (int v = 0; v < 8; ++v) {
                int kl = ks * 32 + ((v < 4) ? (h * 8 + 2 * v) : (16 + h * 8 + 2 * (v - 4)));
                unsigned u0 = __builtin_bit_cast(unsigned, sWc[kl * 32 + ncol]);
                unsigned u1 = __builtin_bit_cast(unsigned, sWc[(kl + 1) * 32 + ncol]);
                // pack {bf16(f1), bf16(f0)} by truncation: one v_perm_b32
                b.u[v] = __builtin_amdgcn_perm(u1, u0, 0x07060302u);
            }

            acc = __builtin_amdgcn_wmma_f32_16x16x32_bf16(
                false, a.bf, false, b.bf, (short)0, acc, false, false);
        }
        __syncthreads();    // reads of buffer `cur` done before it is re-filled
    }

    // C layout: VGPR r -> M = mt*16 + h*8 + r, N = n
    const float bn = bias[n];
#pragma unroll
    for (int r = 0; r < 8; ++r) {
        int m = mt * 16 + h * 8 + r;
        float v = acc[r] + bn;
        if (LEAKY) v = leaky(v);
        if (WRITE_F32)  outF[(size_t)m * N + n] = v;
        if (WRITE_BF16) outB[(size_t)m * N + n] = f32_to_bf16(v);
    }
}

// ---------------------------------------------------------------------------
// Kernel 3: per-sample GCN + head. One block (256 threads) per sample.
// ---------------------------------------------------------------------------
__global__ __launch_bounds__(256)
void gcn_head(const float* __restrict__ nodes_g,   // (B, P*FIN) = (64,12800)
              const int*   __restrict__ ei,        // (B, 2, E)
              const float* __restrict__ Wc1, const float* __restrict__ bc1,
              const float* __restrict__ Wc2, const float* __restrict__ bc2,
              const float* __restrict__ Wf1, const float* __restrict__ bf1,
              const float* __restrict__ Wf2, const float* __restrict__ bf2,
              float* __restrict__ out)             // (B, 1000, 2)
{
    __shared__ __align__(16) float smem[14392];
    float* s_h    = smem;                 // 6400: h1/h2, later aliased by s_z
    float* s_agg  = smem + 6400;          // 6400: agg/x1/x2
    float* s_norm = smem + 12800;         // 450
    float* s_deg  = smem + 13250;         // 50
    int*   s_src  = (int*)(smem + 13300); // 450
    int*   s_dst  = (int*)(smem + 13750); // 450
    float* s_pool = smem + 14200;         // 128
    float* s_zf   = smem + 14328;         // 64
    float* s_z    = smem;                 // 2000 (aliases s_h; h dead by then)

    const int s   = blockIdx.x;
    const int tid = threadIdx.x;
    const float* nodes = nodes_g + (size_t)s * (PP * FIN);
    const int* e_src = ei + (size_t)s * 2 * EE;
    const int* e_dst = e_src + EE;

    // ---- edges (+ self loops) and degree ----
    for (int e = tid; e < NEDGE; e += 256) {
        if (e < EE) { s_src[e] = e_src[e]; s_dst[e] = e_dst[e]; }
        else        { s_src[e] = e - EE;   s_dst[e] = e - EE;  }
    }
    if (tid < PP) s_deg[tid] = 0.f;
    __syncthreads();
    for (int e = tid; e < NEDGE; e += 256)
        atomicAdd(&s_deg[s_dst[e]], 1.0f);        // ds_add_f32
    __syncthreads();
    for (int e = tid; e < NEDGE; e += 256)
        s_norm[e] = rsqrtf(s_deg[s_src[e]]) * rsqrtf(s_deg[s_dst[e]]);
    __syncthreads();

    // ---- h1 = nodes @ Wc1  (50x256 * 256x128) ----
    for (int idx = tid; idx < PP * FH; idx += 256) {
        int p = idx >> 7, j = idx & 127;
        const float* nrow = nodes + p * FIN;
        float acc = 0.f;
        for (int k = 0; k < FIN; ++k)
            acc = fmaf(nrow[k], Wc1[k * FH + j], acc);
        s_h[idx] = acc;
    }
    __syncthreads();

    // ---- aggregate 1: agg[dst] += h[src]*norm ----
    for (int idx = tid; idx < PP * FH; idx += 256) s_agg[idx] = 0.f;
    __syncthreads();
    for (int t = tid; t < NEDGE * FH; t += 256) {
        int e = t >> 7, j = t & 127;
        atomicAdd(&s_agg[s_dst[e] * FH + j], s_h[s_src[e] * FH + j] * s_norm[e]);
    }
    __syncthreads();

    // ---- x1 = leaky(agg + bc1), in place ----
    for (int idx = tid; idx < PP * FH; idx += 256)
        s_agg[idx] = leaky(s_agg[idx] + bc1[idx & 127]);
    __syncthreads();

    // ---- h2 = x1 @ Wc2 (50x128 * 128x128) ----
    for (int idx = tid; idx < PP * FH; idx += 256) {
        int p = idx >> 7, j = idx & 127;
        const float* xrow = s_agg + p * FH;
        float acc = 0.f;
        for (int k = 0; k < FH; ++k)
            acc = fmaf(xrow[k], Wc2[k * FH + j], acc);
        s_h[idx] = acc;
    }
    __syncthreads();

    // ---- aggregate 2 (x1 dead, reuse s_agg) ----
    for (int idx = tid; idx < PP * FH; idx += 256) s_agg[idx] = 0.f;
    __syncthreads();
    for (int t = tid; t < NEDGE * FH; t += 256) {
        int e = t >> 7, j = t & 127;
        atomicAdd(&s_agg[s_dst[e] * FH + j], s_h[s_src[e] * FH + j] * s_norm[e]);
    }
    __syncthreads();

    // ---- x2 = leaky(agg + bc2), in place ----
    for (int idx = tid; idx < PP * FH; idx += 256)
        s_agg[idx] = leaky(s_agg[idx] + bc2[idx & 127]);
    __syncthreads();

    // ---- mean pool over P ----
    if (tid < FH) {
        float acc = 0.f;
        for (int p = 0; p < PP; ++p) acc += s_agg[p * FH + tid];
        s_pool[tid] = acc * (1.0f / (float)PP);
    }
    __syncthreads();

    // ---- zf = leaky(pool @ Wf1 + bf1)  (128 -> 64) ----
    if (tid < HH) {
        float acc = bf1[tid];
        for (int k = 0; k < FH; ++k)
            acc = fmaf(s_pool[k], Wf1[k * HH + tid], acc);
        s_zf[tid] = leaky(acc);
    }
    __syncthreads();

    // ---- z = zf @ Wf2 + bf2  (64 -> 2000); s_z aliases dead s_h ----
    for (int i = tid; i < ZLEN; i += 256) {
        float acc = bf2[i];
        for (int k = 0; k < HH; ++k)
            acc = fmaf(s_zf[k], Wf2[k * ZLEN + i], acc);
        s_z[i] = acc;
    }
    __syncthreads();

    // ---- cumsum over the 1000 axis (2 independent chains) + sigmoid ----
    if (tid < 2) {
        const int c = tid;
        float run = 0.f;
        float* orow = out + (size_t)s * ZLEN;
        for (int i = 0; i < 1000; ++i) {
            run += s_z[i * 2 + c];
            orow[i * 2 + c] = 1.0f / (1.0f + expf(-run));
        }
    }
}

// ---------------------------------------------------------------------------
// Host-side launch
// ---------------------------------------------------------------------------
extern "C" void kernel_launch(void* const* d_in, const int* in_sizes, int n_in,
                              void* d_out, int out_size, void* d_ws, size_t ws_size,
                              hipStream_t stream) {
    const float* b_in = (const float*)d_in[0];   // (64,50,512)
    const int*   ei   = (const int*)  d_in[1];   // (64,2,400)
    const float* W1   = (const float*)d_in[2];
    const float* b1   = (const float*)d_in[3];
    const float* W2   = (const float*)d_in[4];
    const float* b2   = (const float*)d_in[5];
    const float* W3   = (const float*)d_in[6];
    const float* b3   = (const float*)d_in[7];
    const float* Wc1  = (const float*)d_in[8];
    const float* bc1  = (const float*)d_in[9];
    const float* Wc2  = (const float*)d_in[10];
    const float* bc2  = (const float*)d_in[11];
    const float* Wf1  = (const float*)d_in[12];
    const float* bf1  = (const float*)d_in[13];
    const float* Wf2  = (const float*)d_in[14];
    const float* bf2  = (const float*)d_in[15];
    float* out = (float*)d_out;

    // workspace carve-up (all sizes 256B-aligned)
    char* ws = (char*)d_ws;
    unsigned short* xbf   = (unsigned short*)(ws);                          // 64*25600 bf16
    unsigned short* a1bf  = (unsigned short*)(ws + (size_t)BB*K1*2);        // 64*3200 bf16
    unsigned short* a2bf  = (unsigned short*)(ws + (size_t)BB*K1*2
                                                 + (size_t)BB*N1*2);        // 64*6400 bf16
    float*          nodes = (float*)(ws + (size_t)BB*K1*2
                                        + (size_t)BB*N1*2
                                        + (size_t)BB*N2*2);                 // 64*12800 f32

    // 1) convert input activations to bf16
    {
        int n = BB * K1;
        cvt_f32_bf16<<<(n + 255) / 256, 256, 0, stream>>>(b_in, xbf, n);
    }
    // 2) encoder: three TDM-fed bf16-WMMA GEMMs
    gemm_bf16<false, true,  true ><<<N1 / 32, 256, 0, stream>>>(xbf,  W1, b1, nullptr, a1bf, K1, N1);
    gemm_bf16<false, true,  true ><<<N2 / 32, 256, 0, stream>>>(a1bf, W2, b2, nullptr, a2bf, N1, N2);
    gemm_bf16<true,  false, false><<<N3 / 32, 256, 0, stream>>>(a2bf, W3, b3, nodes,  nullptr, N2, N3);
    // 3) per-sample GCN + head + cumsum/sigmoid
    gcn_head<<<BB, 256, 0, stream>>>(nodes, ei, Wc1, bc1, Wc2, bc2,
                                     Wf1, bf1, Wf2, bf2, out);
}
